// Squeeze_Excite_Block_32014686224970
// MI455X (gfx1250) — matrix operand order repaired and checked by reading, not compiled, not run on hardware
//
#include <hip/hip_runtime.h>
#include <hip/hip_bf16.h>
#include <math.h>

// ---------------------------------------------------------------------------
// Types for CDNA5 WMMA (wave32, 16x16x32 bf16 -> f32 accumulate)
// ---------------------------------------------------------------------------
typedef __attribute__((ext_vector_type(16))) __bf16         v16bf;
typedef __attribute__((ext_vector_type(8)))  float          v8f;
typedef __attribute__((ext_vector_type(8)))  unsigned short v8us;
typedef __attribute__((ext_vector_type(16))) unsigned short v16us;

union FragU {
  struct { v8us lo, hi; } p;
  v16us u;
  v16bf b;
};

__device__ __forceinline__ unsigned short f2bf(float f) {
  unsigned u = __float_as_uint(f);
  u += 0x7FFFu + ((u >> 16) & 1u);   // round-to-nearest-even
  return (unsigned short)(u >> 16);
}

__device__ __forceinline__ float interp16(const float* p, int i) {
  // F.interpolate(linear, align_corners=False): half-pixel + edge clamp, 16 -> 128
  float s = (i + 0.5f) * 0.125f - 0.5f;
  int   i0 = (int)floorf(s);
  float fr = s - (float)i0;
  int ia = min(max(i0, 0), 15);
  int ib = min(max(i0 + 1, 0), 15);
  return p[ia] * (1.f - fr) + p[ib] * fr;
}

// Padded y0 geometry: NHWC [8][130][130][128], interior (h,w) at (h+1,w+1)
#define PPIX 16900   // 130*130

// ---------------------------------------------------------------------------
// Stage 1: avg/max pool (3,2,1) concat + BN + Mish -> bf16 padded-NHWC
// ---------------------------------------------------------------------------
__global__ __launch_bounds__(256)
void pool_bn_mish(const float* __restrict__ x, const float* __restrict__ s,
                  const float* __restrict__ b, unsigned short* __restrict__ y0) {
  int idx = blockIdx.x * 256 + threadIdx.x;           // 8*128*16384
  int p = idx & 16383, c = (idx >> 14) & 127, n = idx >> 21;
  int h = p >> 7, w = p & 127;
  float val;
  if (c < 64) {
    const float* src = x + (((size_t)n * 64 + c) << 16);
    float sum = 0.f;
    for (int dy = 0; dy < 3; ++dy) {
      int hh = 2 * h - 1 + dy; if ((unsigned)hh >= 256u) continue;
      for (int dx = 0; dx < 3; ++dx) {
        int ww = 2 * w - 1 + dx; if ((unsigned)ww >= 256u) continue;
        sum += src[(hh << 8) + ww];
      }
    }
    val = sum * (1.f / 9.f);                          // count_include_pad
  } else {
    const float* src = x + (((size_t)n * 64 + (c - 64)) << 16);
    float mx = -INFINITY;
    for (int dy = 0; dy < 3; ++dy) {
      int hh = 2 * h - 1 + dy; if ((unsigned)hh >= 256u) continue;
      for (int dx = 0; dx < 3; ++dx) {
        int ww = 2 * w - 1 + dx; if ((unsigned)ww >= 256u) continue;
        mx = fmaxf(mx, src[(hh << 8) + ww]);
      }
    }
    val = mx;
  }
  val = val * s[c] + b[c];
  float sp = (val > 20.f) ? val : log1pf(expf(val));  // softplus
  val = val * tanhf(sp);                              // Mish
  y0[(((size_t)n * PPIX) + (h + 1) * 130 + (w + 1)) * 128 + c] = f2bf(val);
}

// Zero the 1-pixel halo ring (516 border pixels per image x 128 ch x 8 images)
__global__ __launch_bounds__(256)
void halo_zero(unsigned short* __restrict__ y0) {
  int idx = blockIdx.x * 256 + threadIdx.x;           // 8*516*128
  int c = idx & 127, t = idx >> 7;
  int bi = t % 516, n = t / 516;
  int hp, wp;
  if (bi < 130)      { hp = 0;            wp = bi; }
  else if (bi < 260) { hp = 129;          wp = bi - 130; }
  else if (bi < 388) { hp = bi - 260 + 1; wp = 0; }
  else               { hp = bi - 388 + 1; wp = 129; }
  y0[(((size_t)n * PPIX) + hp * 130 + wp) * 128 + c] = 0;
}

// ---------------------------------------------------------------------------
// Pack weights into per-lane WMMA-B fragment layout:
// dst[((kt*(Ntot/16)+nt)*32 + lane)*16 + e]; lane = ((k>>3)&1)*16 | (o&15),
// e = (k&7) | (((k>>4)&1)<<3) (table order, matching the A fragment loads).
// conv_remap=1: GEMM k is tap-major (k = (ky*3+kx)*128 + c) while the source
// conv weight is [o][c][ky][kx] -> src off = o*K + c*9 + tap.
// ---------------------------------------------------------------------------
__global__ void pack_b(const float* __restrict__ src, unsigned short* __restrict__ dst,
                       int K, int Ntot, int obase, int Nsrc, int conv_remap) {
  int idx = blockIdx.x * 256 + threadIdx.x;
  if (idx >= K * Nsrc) return;
  int osrc = idx / K, k = idx - osrc * K;
  float val;
  if (conv_remap) {
    int tap = k >> 7, c = k & 127;
    val = src[osrc * K + c * 9 + tap];
  } else {
    val = src[idx];
  }
  int o = obase + osrc;
  int kt = k >> 5, nt = o >> 4;
  int lane = (((k >> 3) & 1) << 4) | (o & 15);
  int e = (k & 7) | (((k >> 4) & 1) << 3);
  dst[((((size_t)kt * (Ntot >> 4) + nt) * 32 + lane) << 4) + e] = f2bf(val);
}

__global__ void concat_sb(const float* qs, const float* qb2, const float* ks2, const float* kb2,
                          const float* vs2, const float* vb2, float* sO, float* bO) {
  int t = threadIdx.x;                                // 256
  float s, b;
  if (t < 64)       { s = qs[t];        b = qb2[t]; }
  else if (t < 128) { s = ks2[t - 64];  b = kb2[t - 64]; }
  else              { s = vs2[t - 128]; b = vb2[t - 128]; }
  sO[t] = s; bO[t] = b;
}

// ---------------------------------------------------------------------------
// conv3x3 (128->128, pad 1) as implicit GEMM via WMMA bf16, tap-major K order:
// k = (ky*3+kx)*128 + c, K = 1152.  With the zero-halo padded NHWC input every
// A fragment load is unconditional; the fully unrolled K-loop makes each tap's
// address delta a compile-time immediate, so the loop body is loads + wmma.
// Block: 256 thr = 8 waves, M=32 positions x N=128 out-channels.
// ---------------------------------------------------------------------------
__global__ __launch_bounds__(256)
void conv3x3_wmma(const unsigned short* __restrict__ y0n, const unsigned short* __restrict__ Bp,
                  const float* __restrict__ bias, unsigned short* __restrict__ yn) {
  const int tid = threadIdx.x, l = tid & 31, wv = tid >> 5;
  const int n = blockIdx.y;
  const int p0 = blockIdx.x * 32;
  const int mrow = l & 15, ko = (l >> 4) << 3;
  v8f acc0 = {0.f,0.f,0.f,0.f,0.f,0.f,0.f,0.f};
  v8f acc1 = {0.f,0.f,0.f,0.f,0.f,0.f,0.f,0.f};
  const int p_a = p0 + mrow,      h_a = p_a >> 7, w_a = p_a & 127;
  const int p_b = p0 + 16 + mrow, h_b = p_b >> 7, w_b = p_b & 127;
  // Per-lane base pointers into padded image; tap (ky,kx) adds a *scalar*
  // (compile-time after unroll) offset of ((ky*130+kx)<<7).
  const unsigned short* base = y0n + (size_t)n * (PPIX * 128);
  const unsigned short* va0  = base + ((size_t)(h_a * 130 + w_a) << 7) + ko;
  const unsigned short* va1  = base + ((size_t)(h_b * 130 + w_b) << 7) + ko;
  const unsigned short* bp   = Bp + (((size_t)wv * 32 + l) << 4);
  #pragma unroll
  for (int kt = 0; kt < 36; ++kt) {
    const int tap = kt >> 2, cb = (kt & 3) << 5;
    const int ky = tap / 3, kx = tap - ky * 3;
    const int soff = ((ky * 130 + kx) << 7) + cb;       // constant after unroll
    FragU fa0, fa1, fb;
    fa0.p.lo = *(const v8us*)(va0 + soff);
    fa0.p.hi = *(const v8us*)(va0 + soff + 16);
    fa1.p.lo = *(const v8us*)(va1 + soff);
    fa1.p.hi = *(const v8us*)(va1 + soff + 16);
    fb.u = *(const v16us*)(bp + (size_t)kt * 4096);     // K-step * 8 n-tiles * 512
    acc0 = __builtin_amdgcn_wmma_f32_16x16x32_bf16(false, fa0.b, false, fb.b, (short)0, acc0, false, false);
    acc1 = __builtin_amdgcn_wmma_f32_16x16x32_bf16(false, fa1.b, false, fb.b, (short)0, acc1, false, false);
  }
  int o = wv * 16 + (l & 15);
  float bo = bias[o];
  #pragma unroll
  for (int r = 0; r < 8; ++r) {
    int M = (l < 16) ? r : (8 + r);                   // C/D layout: VGPR r -> M=r / M=8+r
    yn[(((size_t)n * 16384 + (p0 + M))      << 7) + o] = f2bf(acc0[r] + bo);
    yn[(((size_t)n * 16384 + (p0 + 16 + M)) << 7) + o] = f2bf(acc1[r] + bo);
  }
}

// ---------------------------------------------------------------------------
// Generic 1x1-conv GEMM: A = NHWC bf16 [Mtot][K] (contiguous rows; each lane's
// A fragment is two contiguous 16B loads), B pre-packed per-lane fragments.
// EPI 1: qkv split + BN  | EPI 2: pw + BN | EPI 3: proj + BN + hsigmoid*gate.
// ---------------------------------------------------------------------------
template <int K, int NTOT, int EPI>
__global__ __launch_bounds__(256)
void gemm_nhwc(const unsigned short* __restrict__ A, const unsigned short* __restrict__ Bp,
               const float* __restrict__ scale, const float* __restrict__ bias,
               float* __restrict__ o0, float* __restrict__ o1, float* __restrict__ o2,
               const float* __restrict__ gate) {
  const int tid = threadIdx.x, l = tid & 31, wv = tid >> 5;
  const int nt = blockIdx.y * 8 + wv;
  const size_t row0 = (size_t)blockIdx.x * 32;
  v8f acc0 = {0.f,0.f,0.f,0.f,0.f,0.f,0.f,0.f};
  v8f acc1 = {0.f,0.f,0.f,0.f,0.f,0.f,0.f,0.f};
  const int mrow = l & 15, ko = (l >> 4) << 3;
  const unsigned short* a0p = A + (row0 + mrow) * K + ko;
  const unsigned short* a1p = A + (row0 + 16 + mrow) * K + ko;
  const unsigned short* bp  = Bp + (((size_t)nt * 32 + l) << 4);
  #pragma unroll
  for (int kt = 0; kt < K / 32; ++kt) {
    FragU fa0, fa1, fb;
    const v8us* pa = (const v8us*)(a0p + kt * 32);
    fa0.p.lo = pa[0]; fa0.p.hi = pa[2];
    pa = (const v8us*)(a1p + kt * 32);
    fa1.p.lo = pa[0]; fa1.p.hi = pa[2];
    fb.u = *(const v16us*)(bp + (size_t)kt * (NTOT / 16) * 512);
    acc0 = __builtin_amdgcn_wmma_f32_16x16x32_bf16(false, fa0.b, false, fb.b, (short)0, acc0, false, false);
    acc1 = __builtin_amdgcn_wmma_f32_16x16x32_bf16(false, fa1.b, false, fb.b, (short)0, acc1, false, false);
  }
  int o = nt * 16 + (l & 15);
  float sc = scale[o], bi = bias[o];
  #pragma unroll
  for (int r = 0; r < 8; ++r) {
    int M = (l < 16) ? r : (8 + r);
    #pragma unroll
    for (int t = 0; t < 2; ++t) {
      size_t row = row0 + t * 16 + M;
      float v = ((t == 0) ? acc0[r] : acc1[r]) * sc + bi;
      size_t n = row >> 14, p = row & 16383;
      if (EPI == 1) {
        if (o < 64)       o0[((n * 64 + o)          << 14) + p] = v;
        else if (o < 128) o1[((n * 64 + (o - 64))   << 14) + p] = v;
        else              o2[((n * 128 + (o - 128)) << 14) + p] = v;
      } else if (EPI == 2) {
        o0[((n * 128 + o) << 14) + p] = v;
      } else {
        float g = fminf(fmaxf(v + 3.f, 0.f), 6.f) * (1.f / 6.f);   // h_sigmoid
        size_t oi = ((n * 128 + o) << 14) + p;
        o0[oi] = g * gate[oi];
      }
    }
  }
}

// ---------------------------------------------------------------------------
// Depthwise 3x3 over concat(q,k,v) + BN + ReLU -> bf16 NHWC [.,16384,256]
// ---------------------------------------------------------------------------
__global__ __launch_bounds__(256)
void dw_bn_relu(const float* __restrict__ q, const float* __restrict__ kq,
                const float* __restrict__ vq, const float* __restrict__ wdw,
                const float* __restrict__ s, const float* __restrict__ b,
                unsigned short* __restrict__ z) {
  int idx = blockIdx.x * 256 + threadIdx.x;           // 8*256*16384
  int p = idx & 16383, ch = (idx >> 14) & 255, n = idx >> 22;
  int h = p >> 7, w = p & 127;
  const float* src;
  if (ch < 64)       src = q  + (((size_t)n * 64  + ch)         << 14);
  else if (ch < 128) src = kq + (((size_t)n * 64  + (ch - 64))  << 14);
  else               src = vq + (((size_t)n * 128 + (ch - 128)) << 14);
  const float* wt = wdw + ch * 9;
  float sum = 0.f;
  for (int dy = 0; dy < 3; ++dy) {
    int hh = h + dy - 1; if ((unsigned)hh >= 128u) continue;
    for (int dx = 0; dx < 3; ++dx) {
      int ww = w + dx - 1; if ((unsigned)ww >= 128u) continue;
      sum += wt[dy * 3 + dx] * src[(hh << 7) + ww];
    }
  }
  float val = fmaxf(sum * s[ch] + b[ch], 0.f);
  z[(((size_t)n * 16384 + p) << 8) + ch] = f2bf(val);
}

// ---------------------------------------------------------------------------
// Row/col means of q,k,v (+pos interp on q,k) -> out[n][ch 0..255][i]
// ---------------------------------------------------------------------------
__global__ __launch_bounds__(256)
void reduce_mean(const float* __restrict__ q, const float* __restrict__ kq,
                 const float* __restrict__ vq, const float* __restrict__ posq,
                 const float* __restrict__ posk, float* __restrict__ out, int dir) {
  int idx = blockIdx.x * 256 + threadIdx.x;           // 8*256*128
  int i = idx & 127, ch = (idx >> 7) & 255, n = idx >> 15;
  const float* src; const float* pos = nullptr;
  if (ch < 64)       { src = q  + (((size_t)n * 64 + ch) << 14);          pos = posq + ch * 16; }
  else if (ch < 128) { src = kq + (((size_t)n * 64 + (ch - 64)) << 14);   pos = posk + (ch - 64) * 16; }
  else               { src = vq + (((size_t)n * 128 + (ch - 128)) << 14); }
  float sum = 0.f;
  if (dir == 0) { const float* r = src + (i << 7); for (int w = 0; w < 128; ++w) sum += r[w]; }
  else          { const float* r = src + i;        for (int h = 0; h < 128; ++h) sum += r[h << 7]; }
  float val = sum * (1.f / 128.f);
  if (pos) val += interp16(pos, i);
  out[idx] = val;
}

// ---------------------------------------------------------------------------
// Axial attention (per (n,head,i)): softmax_j(q^T k * 8^-0.5) @ v^T
// red layout: [n][ch][len] with q=0..63, k=64..127, v=128..255
// ---------------------------------------------------------------------------
__global__ __launch_bounds__(128)
void axial_attn(const float* __restrict__ red, float* __restrict__ xout) {
  __shared__ float sh[128];
  __shared__ float at[128];
  int j = threadIdx.x;
  int bi = blockIdx.x;                                 // ((n*8 + hd)*128 + i)
  int i = bi & 127, hn = bi >> 7, hd = hn & 7, n = hn >> 3;
  const float* base = red + ((size_t)n << 15);         // n*256*128
  float logit = 0.f;
  #pragma unroll
  for (int kd = 0; kd < 8; ++kd)
    logit += base[(hd * 8 + kd) * 128 + i] * base[(64 + hd * 8 + kd) * 128 + j];
  logit *= 0.35355339059327373f;                       // KD^-0.5
  sh[j] = logit; __syncthreads();
  for (int s2 = 64; s2 > 0; s2 >>= 1) { if (j < s2) sh[j] = fmaxf(sh[j], sh[j + s2]); __syncthreads(); }
  float mx = sh[0]; __syncthreads();
  float e = expf(logit - mx);
  at[j] = e; sh[j] = e; __syncthreads();
  for (int s2 = 64; s2 > 0; s2 >>= 1) { if (j < s2) sh[j] += sh[j + s2]; __syncthreads(); }
  float inv = 1.f / sh[0];
  __syncthreads();
  if (j < 16) {
    const float* vr = base + (128 + hd * 16 + j) * 128;
    float acc = 0.f;
    for (int jj = 0; jj < 128; ++jj) acc += at[jj] * vr[jj];
    xout[((size_t)n * 128 + hd * 16 + j) * 128 + i] = acc * inv;
  }
}

// xout[n][o][i] = s[o]*sum_c w[o][c]*relu(xin[n][c][i]) + b[o]   (tiny GEMM)
__global__ __launch_bounds__(256)
void proj1d(const float* __restrict__ xin, const float* __restrict__ w,
            const float* __restrict__ s, const float* __restrict__ b,
            float* __restrict__ xout) {
  int idx = blockIdx.x * 256 + threadIdx.x;           // 8*128*128
  int i = idx & 127, o = (idx >> 7) & 127, n = idx >> 14;
  const float* xi = xin + (((size_t)n * 128) << 7) + i;
  const float* wr = w + o * 128;
  float acc = 0.f;
  for (int c = 0; c < 128; ++c) acc += wr[c] * fmaxf(xi[c << 7], 0.f);
  xout[idx] = acc * s[o] + b[o];
}

// xx = relu(v + xr(bcast W) + xc(bcast H)) -> bf16 NHWC
__global__ __launch_bounds__(256)
void build_xx(const float* __restrict__ vq, const float* __restrict__ xr,
              const float* __restrict__ xc, unsigned short* __restrict__ xxb) {
  int idx = blockIdx.x * 256 + threadIdx.x;           // 8*128*16384 (NCHW flat)
  int p = idx & 16383, c = (idx >> 14) & 127, n = idx >> 21;
  int h = p >> 7, w = p & 127;
  float val = vq[idx] + xr[((size_t)n * 128 + c) * 128 + h]
                      + xc[((size_t)n * 128 + c) * 128 + w];
  val = fmaxf(val, 0.f);
  xxb[(((size_t)n * 16384 + p) << 7) + c] = f2bf(val);
}

// ---------------------------------------------------------------------------
extern "C" void kernel_launch(void* const* d_in, const int* in_sizes, int n_in,
                              void* d_out, int out_size, void* d_ws, size_t ws_size,
                              hipStream_t stream) {
  const float* x       = (const float*)d_in[0];
  const float* bn1_s   = (const float*)d_in[1];
  const float* bn1_b   = (const float*)d_in[2];
  const float* fc1_w   = (const float*)d_in[3];
  const float* fc1_b   = (const float*)d_in[4];
  const float* wq      = (const float*)d_in[5];
  const float* bnq_s   = (const float*)d_in[6];
  const float* bnq_b   = (const float*)d_in[7];
  const float* wk      = (const float*)d_in[8];
  const float* bnk_s   = (const float*)d_in[9];
  const float* bnk_b   = (const float*)d_in[10];
  const float* wv      = (const float*)d_in[11];
  const float* bnv_s   = (const float*)d_in[12];
  const float* bnv_b   = (const float*)d_in[13];
  const float* pos_rq  = (const float*)d_in[14];
  const float* pos_rk  = (const float*)d_in[15];
  const float* pos_cq  = (const float*)d_in[16];
  const float* pos_ck  = (const float*)d_in[17];
  const float* wrow    = (const float*)d_in[18];
  const float* bnrow_s = (const float*)d_in[19];
  const float* bnrow_b = (const float*)d_in[20];
  const float* wcol    = (const float*)d_in[21];
  const float* bncol_s = (const float*)d_in[22];
  const float* bncol_b = (const float*)d_in[23];
  const float* wproj   = (const float*)d_in[24];
  const float* bnproj_s= (const float*)d_in[25];
  const float* bnproj_b= (const float*)d_in[26];
  const float* wdw     = (const float*)d_in[27];
  const float* bndw_s  = (const float*)d_in[28];
  const float* bndw_b  = (const float*)d_in[29];
  const float* wpw     = (const float*)d_in[30];
  const float* bnpw_s  = (const float*)d_in[31];
  const float* bnpw_b  = (const float*)d_in[32];

  char* ws = (char*)d_ws;
  // Workspace layout (overlays: z over [y0,yn] after both consumed; xxb over q)
  unsigned short* y0   = (unsigned short*)(ws + 0);            // 34611200 B (padded NHWC bf16)
  unsigned short* yn   = (unsigned short*)(ws + 34611200);     // 33554432 B (NHWC bf16)
  unsigned short* z    = (unsigned short*)(ws + 0);            // 67108864 B (overlay)
  float*          qb   = (float*)(ws + 68165632);              // 33554432 B
  unsigned short* xxb  = (unsigned short*)(ws + 68165632);     // 33554432 B (overlay)
  float*          kb   = (float*)(ws + 101720064);             // 33554432 B
  float*          vb   = (float*)(ws + 135274496);             // 67108864 B
  float*          pwo  = (float*)(ws + 202383360);             // 67108864 B
  unsigned short* Bc   = (unsigned short*)(ws + 269492224);    // 294912 B
  unsigned short* Bqkv = (unsigned short*)(ws + 269787136);    // 65536 B
  unsigned short* Bpw  = (unsigned short*)(ws + 269852672);    // 65536 B
  unsigned short* Bprj = (unsigned short*)(ws + 269918208);    // 32768 B
  float*          sQ   = (float*)(ws + 269950976);             // 1024 B
  float*          bQ   = (float*)(ws + 269952000);             // 1024 B
  float*          redr = (float*)(ws + 269953024);             // 1048576 B
  float*          redc = (float*)(ws + 271001600);             // 1048576 B
  float*          xrb  = (float*)(ws + 272050176);             // 524288 B
  float*          xcb  = (float*)(ws + 272574464);             // 524288 B
  float*          xrp  = (float*)(ws + 273098752);             // 524288 B
  float*          xcp  = (float*)(ws + 273623040);             // 524288 B  (total ~261 MB)

  // Weight packing into WMMA-B fragment layouts (w flat as [o][k] = o*K+k;
  // conv uses tap-major remap to match the conv3x3 kernel's K order)
  pack_b<<<(1152 * 128 + 255) / 256, 256, 0, stream>>>(fc1_w, Bc,   1152, 128, 0,   128, 1);
  pack_b<<<(128 * 64 + 255) / 256,  256, 0, stream>>>(wq,    Bqkv, 128,  256, 0,   64,  0);
  pack_b<<<(128 * 64 + 255) / 256,  256, 0, stream>>>(wk,    Bqkv, 128,  256, 64,  64,  0);
  pack_b<<<(128 * 128 + 255) / 256, 256, 0, stream>>>(wv,    Bqkv, 128,  256, 128, 128, 0);
  pack_b<<<(256 * 128 + 255) / 256, 256, 0, stream>>>(wpw,   Bpw,  256,  128, 0,   128, 0);
  pack_b<<<(128 * 128 + 255) / 256, 256, 0, stream>>>(wproj, Bprj, 128,  128, 0,   128, 0);
  concat_sb<<<1, 256, 0, stream>>>(bnq_s, bnq_b, bnk_s, bnk_b, bnv_s, bnv_b, sQ, bQ);

  // Main pipeline
  halo_zero<<<(8 * 516 * 128) / 256, 256, 0, stream>>>(y0);
  pool_bn_mish<<<65536, 256, 0, stream>>>(x, bn1_s, bn1_b, y0);
  conv3x3_wmma<<<dim3(512, 8), 256, 0, stream>>>(y0, Bc, fc1_b, yn);
  gemm_nhwc<128, 256, 1><<<dim3(4096, 2), 256, 0, stream>>>(yn, Bqkv, sQ, bQ, qb, kb, vb, nullptr);
  dw_bn_relu<<<131072, 256, 0, stream>>>(qb, kb, vb, wdw, bndw_s, bndw_b, z);
  gemm_nhwc<256, 128, 2><<<dim3(4096, 1), 256, 0, stream>>>(z, Bpw, bnpw_s, bnpw_b, pwo, nullptr, nullptr, nullptr);
  reduce_mean<<<1024, 256, 0, stream>>>(qb, kb, vb, pos_rq, pos_rk, redr, 0);
  reduce_mean<<<1024, 256, 0, stream>>>(qb, kb, vb, pos_cq, pos_ck, redc, 1);
  axial_attn<<<8192, 128, 0, stream>>>(redr, xrb);
  axial_attn<<<8192, 128, 0, stream>>>(redc, xcb);
  proj1d<<<512, 256, 0, stream>>>(xrb, wrow, bnrow_s, bnrow_b, xrp);
  proj1d<<<512, 256, 0, stream>>>(xcb, wcol, bncol_s, bncol_b, xcp);
  build_xx<<<65536, 256, 0, stream>>>(vb, xrp, xcp, xxb);
  gemm_nhwc<128, 128, 3><<<dim3(4096, 1), 256, 0, stream>>>(xxb, Bprj, bnproj_s, bnproj_b,
                                                            (float*)d_out, nullptr, nullptr, pwo);
  (void)in_sizes; (void)n_in; (void)out_size; (void)ws_size;
}
